// GPTLanguageModel_39917426049408
// MI455X (gfx1250) — compile-verified, matrix operand order checked by reading, not run on hardware
//
#include <hip/hip_runtime.h>

// ---------------------------------------------------------------------------
// GPT forward on gfx1250 (MI455X): bf16 WMMA GEMMs + fused LDS attention.
// Round 4: fix async-builtin pointer element type (int vector_size(16), per
// hipcc diagnostic). A-operand staged via GLOBAL_LOAD_ASYNC_TO_LDS_B128
// (ASYNCcnt path); B-operand register staging (K-major -> N-major transpose).
// ---------------------------------------------------------------------------

#define Vv_ 32000
#define Cc_ 1024
#define Tt_ 1024
#define Hh_ 16
#define HSs_ 64
#define Ll_ 8
#define Bb_ 4
#define MM_ 4096   // B*T

typedef __attribute__((ext_vector_type(16))) __bf16 v16bf;
typedef __attribute__((ext_vector_type(8)))  float  v8f;
typedef int v4i_ __attribute__((__vector_size__(16)));   // matches builtin proto

#define AS1_ __attribute__((address_space(1)))
#define AS3_ __attribute__((address_space(3)))

union Frag { v16bf v; __bf16 s[16]; };

__device__ __forceinline__ __bf16 f2bf(float f) { return (__bf16)f; }
__device__ __forceinline__ float bf2f(__bf16 h) { return (float)h; }

__device__ __forceinline__ v8f wmma_bf16(v16bf a, v16bf b, v8f c) {
  return __builtin_amdgcn_wmma_f32_16x16x32_bf16(
      /*neg_a=*/false, a, /*neg_b=*/false, b,
      /*c_mod=*/(short)0, c, /*reuse_a=*/false, /*reuse_b=*/false);
}

// 16-byte global -> LDS copy: async DMA if the toolchain exposes it.
__device__ __forceinline__ void copy16_g2l(const float* g, float* l) {
#if __has_builtin(__builtin_amdgcn_global_load_async_to_lds_b128)
  __builtin_amdgcn_global_load_async_to_lds_b128(
      (AS1_ v4i_*)(g), (AS3_ v4i_*)(l), 0, 0);
#else
  *(float4*)l = *(const float4*)g;
#endif
}
__device__ __forceinline__ void wait_async_copies() {
#if __has_builtin(__builtin_amdgcn_global_load_async_to_lds_b128)
#if __has_builtin(__builtin_amdgcn_s_wait_asynccnt)
  __builtin_amdgcn_s_wait_asynccnt(0);
#else
  asm volatile("s_wait_asynccnt 0x0" ::: "memory");
#endif
#endif
}

// ---------------------------------------------------------------------------
// Embedding: x[b,t,:] = tok_emb[index[b,t],:] + pos_emb[t,:]
// ---------------------------------------------------------------------------
__global__ __launch_bounds__(256) void embed_kernel(
    const int* __restrict__ idx, const float* __restrict__ tok,
    const float* __restrict__ pos, float* __restrict__ x) {
  int row = blockIdx.x;                 // 0..4095
  int t = row & (Tt_ - 1);
  int v = idx[row];
  const float* tp = tok + (size_t)v * Cc_;
  const float* pp = pos + (size_t)t * Cc_;
  float* xp = x + (size_t)row * Cc_;
  for (int c = threadIdx.x; c < Cc_; c += 256) xp[c] = tp[c] + pp[c];
}

// ---------------------------------------------------------------------------
// GEMM: out[M,N] = act( A[M,K] @ W[K,N] + bias + resid ), bf16 WMMA, fp32 acc.
// Block tile 128x64, 256 threads = 8 waves (4x2); wave tile 32x32 = 4 WMMAs
// per K-step. Ping-pong LDS buffers, one barrier per K-step. A staged by
// async global->LDS DMA (fp32, converted at fragment read); B staged through
// registers with bf16 convert + transpose.
// ---------------------------------------------------------------------------
__global__ __launch_bounds__(256) void gemm_kernel(
    const float* __restrict__ A, const float* __restrict__ W,
    const float* __restrict__ bias, const float* __restrict__ resid,
    float* __restrict__ out, int Mm, int Nn, int Kk, int relu) {
  __shared__ float  Af[2][128][36];   // [buf][m][k]  fp32 (async DMA target)
  __shared__ __bf16 Bs[2][64][36];    // [buf][n][k]  bf16 (transposed staging)
  const int tid = threadIdx.x;
  const int lane = tid & 31;
  const int wave = tid >> 5;
  const int n0 = blockIdx.x * 64;
  const int m0 = blockIdx.y * 128;
  const int wm = wave >> 1;           // 0..3  -> 32-row strip
  const int wn = wave & 1;            // 0..1  -> 32-col strip

  // staging assignments
  const int a_mm = tid >> 1;          // 0..127
  const int a_kb = (tid & 1) * 16;    // 0 or 16
  const int b_kk = tid >> 3;          // 0..31
  const int b_nb = (tid & 7) * 8;     // 0..56

  const float* a_src = A + (size_t)(m0 + a_mm) * Kk + a_kb;
  const float* b_src = W + (size_t)b_kk * Nn + n0 + b_nb;

  auto stage_A = [&](int k0, int buf) {     // 4x16B per thread, async if avail
    const float* g = a_src + k0;
    float* l = &Af[buf][a_mm][a_kb];
#pragma unroll
    for (int i = 0; i < 4; i++) copy16_g2l(g + 4 * i, l + 4 * i);
  };
  auto load_B = [&](int k0, float4* br) {
    const float4* bp = (const float4*)(b_src + (size_t)k0 * Nn);
    br[0] = bp[0]; br[1] = bp[1];
  };
  auto store_B = [&](int buf, const float4* br) {
    const float* bf = (const float*)br;
#pragma unroll
    for (int i = 0; i < 8; i++) Bs[buf][b_nb + i][b_kk] = f2bf(bf[i]);
  };

  v8f acc[2][2] = {{{}, {}}, {{}, {}}};

  // prologue: stage K-step 0 into buffer 0
  {
    float4 br[2];
    stage_A(0, 0);
    load_B(0, br);
    store_B(0, br);
  }
  wait_async_copies();
  __syncthreads();

  const int kSteps = Kk >> 5;
  for (int s = 0; s < kSteps; ++s) {
    const int buf = s & 1;
    const bool more = (s + 1) < kSteps;

    float4 br[2];
    if (more) {
      stage_A((s + 1) << 5, buf ^ 1);                    // async DMA -> LDS
      load_B((s + 1) << 5, br);                          // global -> regs
    }
    if (s + 2 < kSteps) {                                // warm GL2 two steps out
      __builtin_prefetch(a_src + ((s + 2) << 5), 0, 1);
      __builtin_prefetch(b_src + ((size_t)((s + 2) << 5)) * Nn, 0, 1);
    }

    // compute from LDS[buf] (A converted fp32->bf16 at fragment read)
    Frag fa[2], fb[2];
    {
      const int lr = lane & 15;
      const int kb = (lane >> 4) * 8;
#pragma unroll
      for (int i = 0; i < 2; i++) {
        int r = wm * 32 + i * 16 + lr;
#pragma unroll
        for (int j = 0; j < 8; j++) {
          fa[i].s[j] = f2bf(Af[buf][r][kb + j]);
          fa[i].s[8 + j] = f2bf(Af[buf][r][kb + 16 + j]);
        }
      }
#pragma unroll
      for (int i = 0; i < 2; i++) {
        int r = wn * 32 + i * 16 + lr;
#pragma unroll
        for (int j = 0; j < 8; j++) {
          fb[i].s[j] = Bs[buf][r][kb + j];
          fb[i].s[8 + j] = Bs[buf][r][kb + 16 + j];
        }
      }
    }
    acc[0][0] = wmma_bf16(fa[0].v, fb[0].v, acc[0][0]);
    acc[0][1] = wmma_bf16(fa[0].v, fb[1].v, acc[0][1]);
    acc[1][0] = wmma_bf16(fa[1].v, fb[0].v, acc[1][0]);
    acc[1][1] = wmma_bf16(fa[1].v, fb[1].v, acc[1][1]);

    if (more) store_B(buf ^ 1, br);                      // fill other buffer
    wait_async_copies();                                 // own DMAs done
    __syncthreads();                                     // everyone's done
  }

  // epilogue: D layout -> (m = 8*(lane>>4)+v, n = lane&15)
  const int mb = (lane >> 4) * 8;
  const int nl = lane & 15;
#pragma unroll
  for (int i = 0; i < 2; i++) {
#pragma unroll
    for (int j = 0; j < 2; j++) {
      int col = n0 + wn * 32 + j * 16 + nl;
      float bia = bias ? bias[col] : 0.f;
#pragma unroll
      for (int vv = 0; vv < 8; vv++) {
        int row = m0 + wm * 32 + i * 16 + mb + vv;
        float val = acc[i][j][vv] + bia;
        if (resid) val += resid[(size_t)row * Nn + col];
        if (relu) val = fmaxf(val, 0.f);
        out[(size_t)row * Nn + col] = val;
      }
    }
  }
}

// ---------------------------------------------------------------------------
// Fused causal attention for one (b, h, 16-query tile).
// Scores strip (16 x T) kept in LDS as bf16; QK^T and PV both use WMMA bf16.
// ---------------------------------------------------------------------------
__global__ __launch_bounds__(256) void attn_kernel(
    const float* __restrict__ Q, const float* __restrict__ K,
    const float* __restrict__ V, float* __restrict__ O) {
  const int tid = threadIdx.x;
  const int lane = tid & 31;
  const int wave = tid >> 5;
  const int qt = blockIdx.x & 63;               // T/16
  const int hh = (blockIdx.x >> 6) & (Hh_ - 1);
  const int b  = blockIdx.x >> 10;
  const float scale = 0.125f;                   // 1/sqrt(64)

  __shared__ __bf16 sc[16][1040];               // bf16 scores/probs
  __shared__ float red[16][16];
  __shared__ float rowsum[16];

  // ---- load Q fragments (16x64 tile -> two 16x32 bf16 frags), all waves ----
  Frag aq0, aq1;
  {
    int m = lane & 15;
    int kb = (lane >> 4) * 8;
    int tq = qt * 16 + m;
    const float* qp = Q + ((size_t)(b * Tt_ + tq)) * Cc_ + hh * HSs_;
#pragma unroll
    for (int j = 0; j < 8; j++) {
      aq0.s[j]     = f2bf(qp[kb + j]);
      aq0.s[8 + j] = f2bf(qp[kb + 16 + j]);
      aq1.s[j]     = f2bf(qp[32 + kb + j]);
      aq1.s[8 + j] = f2bf(qp[32 + kb + 16 + j]);
    }
  }

  // ---- phase 1: scores = Q @ K^T, waves split key tiles round-robin ----
  for (int kt = wave; kt <= qt; kt += 8) {
    Frag bk0, bk1;
    int n = lane & 15;
    int kb = (lane >> 4) * 8;
    int tk = kt * 16 + n;
    const float* kp = K + ((size_t)(b * Tt_ + tk)) * Cc_ + hh * HSs_;
#pragma unroll
    for (int j = 0; j < 8; j++) {
      bk0.s[j]     = f2bf(kp[kb + j]);
      bk0.s[8 + j] = f2bf(kp[kb + 16 + j]);
      bk1.s[j]     = f2bf(kp[32 + kb + j]);
      bk1.s[8 + j] = f2bf(kp[32 + kb + 16 + j]);
    }
    v8f acc = {};
    acc = wmma_bf16(aq0.v, bk0.v, acc);
    acc = wmma_bf16(aq1.v, bk1.v, acc);

    int kcol = kt * 16 + (lane & 15);
    int mb = (lane >> 4) * 8;
#pragma unroll
    for (int vv = 0; vv < 8; vv++) {
      int qg = qt * 16 + mb + vv;
      float val = (kcol <= qg) ? acc[vv] * scale : -INFINITY;
      sc[mb + vv][kcol] = f2bf(val);
    }
  }
  __syncthreads();

  // ---- phase 2: softmax over each of the 16 rows (16 threads per row) ----
  {
    int row = tid >> 4;
    int sub = tid & 15;
    int len = (qt + 1) * 16;
    float mx = -INFINITY;
    for (int c = sub; c < len; c += 16) mx = fmaxf(mx, bf2f(sc[row][c]));
    red[row][sub] = mx;
    __syncthreads();
#pragma unroll
    for (int i = 0; i < 16; i++) mx = fmaxf(mx, red[row][i]);
    __syncthreads();
    float s = 0.f;
    for (int c = sub; c < len; c += 16) {
      float p = __expf(bf2f(sc[row][c]) - mx);
      sc[row][c] = f2bf(p);
      s += p;
    }
    red[row][sub] = s;
    __syncthreads();
    s = 0.f;
#pragma unroll
    for (int i = 0; i < 16; i++) s += red[row][i];
    if (sub == 0) rowsum[row] = s;
    // zero-pad probs to a multiple of 32 key columns for the PV WMMA loop
    int pad = ((qt + 2) & ~1) * 16;
    for (int c = len + sub; c < pad; c += 16) sc[row][c] = f2bf(0.f);
  }
  __syncthreads();

  // ---- phase 3: out = P @ V, waves 0..3 each own a 16-wide d-subtile ----
  if (wave < 4) {
    const int ns = wave;
    const int n = lane & 15;
    const int kb = (lane >> 4) * 8;
    const int d = ns * 16 + n;
    const int m = lane & 15;
    const int pad = ((qt + 2) & ~1) * 16;
    const float* vbase = V + ((size_t)(b * Tt_)) * Cc_ + hh * HSs_ + d;
    v8f acc = {};
    for (int kc = 0; kc < pad; kc += 32) {
      Frag ap, bv;
#pragma unroll
      for (int j = 0; j < 8; j++) {
        ap.s[j]     = sc[m][kc + kb + j];
        ap.s[8 + j] = sc[m][kc + kb + 16 + j];
      }
#pragma unroll
      for (int j = 0; j < 8; j++) {
        bv.s[j]     = f2bf(vbase[(size_t)(kc + kb + j) * Cc_]);
        bv.s[8 + j] = f2bf(vbase[(size_t)(kc + kb + 16 + j) * Cc_]);
      }
      acc = wmma_bf16(ap.v, bv.v, acc);
    }
    int mb = (lane >> 4) * 8;
#pragma unroll
    for (int vv = 0; vv < 8; vv++) {
      int tq = qt * 16 + mb + vv;
      float o = acc[vv] / rowsum[mb + vv];
      O[((size_t)(b * Tt_ + tq)) * Cc_ + hh * HSs_ + d] = o;
    }
  }
}

// ---------------------------------------------------------------------------
// LayerNorm over C=1024 per row.
// ---------------------------------------------------------------------------
__global__ __launch_bounds__(256) void ln_kernel(
    const float* __restrict__ in, const float* __restrict__ g,
    const float* __restrict__ bta, float* __restrict__ out) {
  __shared__ float sbuf[256];
  int row = blockIdx.x, tid = threadIdx.x;
  const float* ip = in + (size_t)row * Cc_;
  float s = 0.f;
  for (int c = tid; c < Cc_; c += 256) s += ip[c];
  sbuf[tid] = s; __syncthreads();
  for (int o = 128; o > 0; o >>= 1) { if (tid < o) sbuf[tid] += sbuf[tid + o]; __syncthreads(); }
  float mean = sbuf[0] * (1.f / Cc_);
  __syncthreads();
  float vs = 0.f;
  for (int c = tid; c < Cc_; c += 256) { float d = ip[c] - mean; vs += d * d; }
  sbuf[tid] = vs; __syncthreads();
  for (int o = 128; o > 0; o >>= 1) { if (tid < o) sbuf[tid] += sbuf[tid + o]; __syncthreads(); }
  float inv = rsqrtf(sbuf[0] * (1.f / Cc_) + 1e-5f);
  float* op = out + (size_t)row * Cc_;
  for (int c = tid; c < Cc_; c += 256) op[c] = (ip[c] - mean) * inv * g[c] + bta[c];
}

// ---------------------------------------------------------------------------
// Per-row cross-entropy: rowloss[r] = log(sum exp(l)) + max - l[target]
// ---------------------------------------------------------------------------
__global__ __launch_bounds__(256) void loss_row_kernel(
    const float* __restrict__ logits, const int* __restrict__ targets,
    float* __restrict__ rowloss) {
  __shared__ float sbuf[256];
  int row = blockIdx.x, tid = threadIdx.x;
  const float* lp = logits + (size_t)row * Vv_;
  float mx = -INFINITY;
  for (int c = tid; c < Vv_; c += 256) mx = fmaxf(mx, lp[c]);
  sbuf[tid] = mx; __syncthreads();
  for (int o = 128; o > 0; o >>= 1) { if (tid < o) sbuf[tid] = fmaxf(sbuf[tid], sbuf[tid + o]); __syncthreads(); }
  mx = sbuf[0];
  __syncthreads();
  float s = 0.f;
  for (int c = tid; c < Vv_; c += 256) s += __expf(lp[c] - mx);
  sbuf[tid] = s; __syncthreads();
  for (int o = 128; o > 0; o >>= 1) { if (tid < o) sbuf[tid] += sbuf[tid + o]; __syncthreads(); }
  if (tid == 0) rowloss[row] = (__logf(sbuf[0]) + mx) - lp[targets[row]];
}

__global__ __launch_bounds__(256) void loss_final_kernel(
    const float* __restrict__ rowloss, float* __restrict__ out) {
  __shared__ float sbuf[256];
  int tid = threadIdx.x;
  float s = 0.f;
  for (int i = tid; i < MM_; i += 256) s += rowloss[i];
  sbuf[tid] = s; __syncthreads();
  for (int o = 128; o > 0; o >>= 1) { if (tid < o) sbuf[tid] += sbuf[tid + o]; __syncthreads(); }
  if (tid == 0) out[0] = sbuf[0] * (1.f / MM_);
}

// ---------------------------------------------------------------------------
// Host orchestration.
// ---------------------------------------------------------------------------
extern "C" void kernel_launch(void* const* d_in, const int* in_sizes, int n_in,
                              void* d_out, int out_size, void* d_ws, size_t ws_size,
                              hipStream_t stream) {
  (void)in_sizes; (void)n_in; (void)out_size; (void)ws_size;
  const int*   index   = (const int*)  d_in[0];
  const int*   targets = (const int*)  d_in[1];
  const float* tok_emb = (const float*)d_in[2];
  const float* pos_emb = (const float*)d_in[3];
  const float* wq      = (const float*)d_in[4];
  const float* wk      = (const float*)d_in[5];
  const float* wv      = (const float*)d_in[6];
  const float* wo      = (const float*)d_in[7];
  const float* bo      = (const float*)d_in[8];
  const float* w1      = (const float*)d_in[9];
  const float* b1      = (const float*)d_in[10];
  const float* w2      = (const float*)d_in[11];
  const float* b2      = (const float*)d_in[12];
  const float* ln1_g   = (const float*)d_in[13];
  const float* ln1_b   = (const float*)d_in[14];
  const float* ln2_g   = (const float*)d_in[15];
  const float* ln2_b   = (const float*)d_in[16];
  const float* lnf_g   = (const float*)d_in[17];
  const float* lnf_b   = (const float*)d_in[18];
  const float* lm_w    = (const float*)d_in[19];
  const float* lm_b    = (const float*)d_in[20];

  float* logits = (float*)d_out;                       // [4096, 32000]
  float* loss   = logits + (size_t)MM_ * Vv_;          // [1]

  // workspace layout (floats)
  float* ws  = (float*)d_ws;
  float* x   = ws;                                     // [4096,1024]
  float* t0  = x   + (size_t)MM_ * Cc_;                // [4096,1024]
  float* q   = t0  + (size_t)MM_ * Cc_;                // [4096,1024]
  float* k   = q   + (size_t)MM_ * Cc_;                // [4096,1024]
  float* v   = k   + (size_t)MM_ * Cc_;                // [4096,1024]
  float* att = v   + (size_t)MM_ * Cc_;                // [4096,1024]
  float* h   = att + (size_t)MM_ * Cc_;                // [4096,4096]
  float* rls = h   + (size_t)MM_ * (4 * Cc_);          // [4096]

  const dim3 blk(256);
  const dim3 gC(Cc_ / 64, MM_ / 128);                  // N=1024 GEMMs
  const dim3 gH(4 * Cc_ / 64, MM_ / 128);              // N=4096 GEMM
  const dim3 gV(Vv_ / 64, MM_ / 128);                  // lm_head

  embed_kernel<<<MM_, blk, 0, stream>>>(index, tok_emb, pos_emb, x);

  for (int l = 0; l < Ll_; l++) {
    const float* wql = wq + (size_t)l * Cc_ * Cc_;
    const float* wkl = wk + (size_t)l * Cc_ * Cc_;
    const float* wvl = wv + (size_t)l * Cc_ * Cc_;
    const float* wol = wo + (size_t)l * Cc_ * Cc_;
    const float* w1l = w1 + (size_t)l * Cc_ * 4 * Cc_;
    const float* w2l = w2 + (size_t)l * 4 * Cc_ * Cc_;

    gemm_kernel<<<gC, blk, 0, stream>>>(x, wql, nullptr, nullptr, q, MM_, Cc_, Cc_, 0);
    gemm_kernel<<<gC, blk, 0, stream>>>(x, wkl, nullptr, nullptr, k, MM_, Cc_, Cc_, 0);
    gemm_kernel<<<gC, blk, 0, stream>>>(x, wvl, nullptr, nullptr, v, MM_, Cc_, Cc_, 0);

    attn_kernel<<<Bb_ * Hh_ * (Tt_ / 16), blk, 0, stream>>>(q, k, v, att);

    // t0 = x + att @ wo + bo ; x = LN(t0)
    gemm_kernel<<<gC, blk, 0, stream>>>(att, wol, bo + (size_t)l * Cc_, x, t0, MM_, Cc_, Cc_, 0);
    ln_kernel<<<MM_, blk, 0, stream>>>(t0, ln1_g + (size_t)l * Cc_, ln1_b + (size_t)l * Cc_, x);

    // h = relu(x @ w1 + b1) ; t0 = x + h @ w2 + b2 ; x = LN(t0)
    gemm_kernel<<<gH, blk, 0, stream>>>(x, w1l, b1 + (size_t)l * 4 * Cc_, nullptr, h, MM_, 4 * Cc_, Cc_, 1);
    gemm_kernel<<<gC, blk, 0, stream>>>(h, w2l, b2 + (size_t)l * Cc_, x, t0, MM_, Cc_, 4 * Cc_, 0);
    ln_kernel<<<MM_, blk, 0, stream>>>(t0, ln2_g + (size_t)l * Cc_, ln2_b + (size_t)l * Cc_, x);
  }

  // final LN + lm_head
  ln_kernel<<<MM_, blk, 0, stream>>>(x, lnf_g, lnf_b, t0);
  gemm_kernel<<<gV, blk, 0, stream>>>(t0, lm_w, lm_b, nullptr, logits, MM_, Vv_, Cc_, 0);

  // loss
  loss_row_kernel<<<MM_, blk, 0, stream>>>(logits, targets, rls);
  loss_final_kernel<<<1, blk, 0, stream>>>(rls, loss);
}
